// SSCDNonLModel_gcn3_90374701843195
// MI455X (gfx1250) — matrix operand order repaired and checked by reading, not compiled, not run on hardware
//
#include <hip/hip_runtime.h>
#include <math.h>

#define NP 6400
#define NPT 400            // pixel tiles of 16
#define HH 80
#define WWI 80
#define BN_EPS 1e-5f
#define THRESHV 2e-4f

typedef __attribute__((ext_vector_type(16))) _Float16 v16h;
typedef __attribute__((ext_vector_type(8)))  float    v8f;

__device__ __forceinline__ float sigmoidf_(float v){ return 1.0f/(1.0f+__expf(-v)); }

// ---------- fold eval-mode BN (on conv input channels) into 1x1 conv ----------
__global__ void fold_bn_conv(const float* __restrict__ Wc, const float* __restrict__ bc,
                             const float* __restrict__ g,  const float* __restrict__ bb,
                             const float* __restrict__ mn, const float* __restrict__ vr,
                             int O, int C, float* __restrict__ Wf, float* __restrict__ bf){
  int o = blockIdx.x*blockDim.x + threadIdx.x;
  if(o >= O) return;
  float acc = bc[o];
  for(int c=0;c<C;c++){
    float scale = g[c]*rsqrtf(vr[c]+BN_EPS);
    float shift = bb[c]-mn[c]*scale;
    float w = Wc[o*C+c];
    Wf[o*C+c] = w*scale;
    acc += w*shift;
  }
  bf[o] = acc;
}

// ---------- pack f32 weights into WMMA A-fragment f16 layout ----------
// AW index = ((mt*nkc + kc)*32 + lane)*16 + j ; zero-padded in M and K
// A 16x32 (MxK): M = lane&15; lanes 0-15 hold K j<8?j:j+8, lanes 16-31 hold j<8?j+8:j+16
__global__ void pack_w_f16(const float* __restrict__ Wf, _Float16* __restrict__ AW,
                           int M, int K, int nkc){
  int idx = blockIdx.x*blockDim.x + threadIdx.x;
  int total = ((M+15)/16)*nkc*512;
  if(idx >= total) return;
  int j    = idx & 15;
  int lane = (idx>>4) & 31;
  int kc   = (idx>>9) % nkc;
  int mt   = (idx>>9) / nkc;
  int m = mt*16 + (lane & 15);
  int ka = (lane<16) ? (j<8 ? j : j+8) : (j<8 ? j+8 : j+16);
  int k = kc*32 + ka;
  AW[idx] = (m<M && k<K) ? (_Float16)Wf[m*K+k] : (_Float16)0.0f;
}

// ---------- pack f32 activations [K][NP] into WMMA B-fragment f16 layout ----------
// BX index = ((kc*NPT + pt)*32 + lane)*16 + j ; zero-padded in K
// B 32x16 (KxN): N = lane&15; lanes 0-15 hold K j<8?j:j+8, lanes 16-31 hold j<8?j+8:j+16
__global__ void pack_x_f16(const float* __restrict__ X, _Float16* __restrict__ BX,
                           int K, int nkc){
  int idx = blockIdx.x*blockDim.x + threadIdx.x;
  int total = nkc*NPT*512;
  if(idx >= total) return;
  int j    = idx & 15;
  int lane = (idx>>4) & 31;
  int pt   = (idx>>9) % NPT;
  int kc   = (idx>>9) / NPT;
  int k = kc*32 + ((lane<16) ? (j<8 ? j : j+8) : (j<8 ? j+8 : j+16));
  int n = pt*16 + (lane & 15);
  BX[idx] = (k<K) ? (_Float16)X[(size_t)k*NP + n] : (_Float16)0.0f;
}

// ---------- WMMA f16 GEMM on pre-packed fragments ----------
// out[M][NP] = act(W @ X + b); one wave32 per 16x16 tile; branch-free inner loop
__global__ void gemm_wmma(const _Float16* __restrict__ AW, const _Float16* __restrict__ BX,
                          const float* __restrict__ bias, float* __restrict__ out,
                          int M, int nkc, int act){
  int lane = threadIdx.x;
  int ptile = blockIdx.x*blockDim.y + threadIdx.y;   // 0..399
  int mt = blockIdx.y;
  int ml = lane & 15;
  v8f acc = {};
  const _Float16* ap = AW + (((size_t)mt*nkc)*32 + (size_t)lane)*16;
  const _Float16* bp = BX + (((size_t)ptile)*32 + (size_t)lane)*16;
  #pragma unroll 2
  for(int kc=0; kc<nkc; kc++){
    v16h a = *(const v16h*)(ap + (size_t)kc*512);
    v16h b = *(const v16h*)(bp + (size_t)kc*NPT*512);
    acc = __builtin_amdgcn_wmma_f32_16x16x32_f16(false, a, false, b, (short)0, acc, false, false);
  }
  int pix0 = ptile*16;
  #pragma unroll
  for(int v=0; v<8; v++){
    int m = mt*16 + v + ((lane<16)?0:8);             // C/D: lanes 0-15 M=0..7, 16-31 M=8..15
    if(m < M){
      float val = acc[v] + bias[m];
      if(act) val = sigmoidf_(val);
      out[(size_t)m*NP + pix0 + ml] = val;
    }
  }
}

// ---------- depthwise / grouped 5x5 conv, pad=2, sigmoid ----------
__global__ void dwconv5x5_sig(const float* __restrict__ X, const float* __restrict__ Wt,
                              const float* __restrict__ bias, float* __restrict__ out,
                              int O, int Gin){
  int idx = blockIdx.x*blockDim.x + threadIdx.x;
  if(idx >= O*NP) return;
  int o = idx/NP, p = idx%NP;
  int y = p/WWI, x = p%WWI;
  float acc = bias[o];
  for(int gch=0; gch<Gin; gch++){
    int cin = o*Gin + gch;
    const float* xp = X + (size_t)cin*NP;
    const float* wp = Wt + (o*Gin+gch)*25;
    #pragma unroll
    for(int ky=0;ky<5;ky++){
      int yy = y+ky-2;
      if(yy<0||yy>=HH) continue;
      #pragma unroll
      for(int kx=0;kx<5;kx++){
        int xx = x+kx-2;
        if(xx<0||xx>=WWI) continue;
        acc += xp[yy*WWI+xx]*wp[ky*5+kx];
      }
    }
  }
  out[idx] = sigmoidf_(acc);
}

// ---------- elementwise eval BN ----------
__global__ void bn_elem(const float* __restrict__ X, const float* g, const float* bb,
                        const float* mn, const float* vr, float* out, int C){
  int idx = blockIdx.x*blockDim.x + threadIdx.x;
  if(idx >= C*NP) return;
  int c = idx/NP;
  float scale = g[c]*rsqrtf(vr[c]+BN_EPS);
  out[idx] = (X[idx]-mn[c])*scale + bb[c];
}

// ---------- per-row online-softmax stats -> log-domain threshold t[i] ----------
// att[i,j] >= THRESH  <=>  e[i,j] >= max_i + log(THRESH * sumexp_i)
__global__ void row_stats(const float* __restrict__ q, const float* __restrict__ k,
                          float* __restrict__ tthr){
  int row = blockIdx.x*blockDim.y + threadIdx.y;
  int lane = threadIdx.x;
  float q0=q[row], q1=q[NP+row], q2=q[2*NP+row];
  float m=-1e30f, s=0.0f;
  for(int j=lane;j<NP;j+=32){
    float e = q0*k[j] + q1*k[NP+j] + q2*k[2*NP+j];
    if(e<=m) s += __expf(e-m);
    else { s = s*__expf(m-e) + 1.0f; m = e; }
  }
  for(int off=16; off>0; off>>=1){
    float m2 = __shfl_xor(m,off);
    float s2 = __shfl_xor(s,off);
    float mm = fmaxf(m,m2);
    s = s*__expf(m-mm) + s2*__expf(m2-mm);
    m = mm;
  }
  if(lane==0) tthr[row] = m + __logf(THRESHV*s);
}

// ---------- per-column degree of thresholded adjacency (diag forced 1) ----------
__global__ void col_deg(const float* __restrict__ q, const float* __restrict__ k,
                        const float* __restrict__ tthr, float* __restrict__ dinv){
  int col = blockIdx.x*blockDim.y + threadIdx.y;
  int lane = threadIdx.x;
  float k0=k[col], k1=k[NP+col], k2=k[2*NP+col];
  float cnt = 0.0f;
  for(int i=lane;i<NP;i+=32){
    float e = q[i]*k0 + q[NP+i]*k1 + q[2*NP+i]*k2;
    cnt += ((i==col) || (e>=tthr[i])) ? 1.0f : 0.0f;
  }
  for(int off=16; off>0; off>>=1) cnt += __shfl_xor(cnt,off);
  if(lane==0) dinv[col] = rsqrtf(cnt);
}

// ---------- GCN layer1 is rank-1: s[j] = sum_i A[i,j]*dinv[i]*xt[i] ----------
__global__ void col_sacc(const float* __restrict__ q, const float* __restrict__ k,
                         const float* __restrict__ tthr, const float* __restrict__ dinv,
                         const float* __restrict__ xt, float* __restrict__ sout){
  int col = blockIdx.x*blockDim.y + threadIdx.y;
  int lane = threadIdx.x;
  float k0=k[col], k1=k[NP+col], k2=k[2*NP+col];
  float acc = 0.0f;
  for(int i=lane;i<NP;i+=32){
    float e = q[i]*k0 + q[NP+i]*k1 + q[2*NP+i]*k2;
    if((i==col) || (e>=tthr[i])) acc += dinv[i]*xt[i];
  }
  for(int off=16; off>0; off>>=1) acc += __shfl_xor(acc,off);
  if(lane==0) sout[col] = acc;
}

// ---------- feat2d[i][c] = dinv[i] * (relu(dinv_i*s_i*w1 + b1) @ W2)[c] ----------
__global__ void feat2_build(const float* __restrict__ dinv, const float* __restrict__ s,
                            const float* __restrict__ w1, const float* __restrict__ b1,
                            const float* __restrict__ W2, float* __restrict__ feat2d){
  int i = blockIdx.x*blockDim.x + threadIdx.x;
  if(i >= NP) return;
  float sc = dinv[i]*s[i];
  float f2[25];
  #pragma unroll
  for(int c=0;c<25;c++) f2[c]=0.0f;
  for(int f=0;f<12;f++){
    float g1f = fmaxf(sc*w1[f]+b1[f], 0.0f);
    #pragma unroll
    for(int c=0;c<25;c++) f2[c] += g1f*W2[f*25+c];
  }
  float di = dinv[i];
  for(int c=0;c<25;c++) feat2d[i*25+c] = di*f2[c];
}

// ---------- GCN layer2 masked aggregation: g2[j,c] = dinv_j*sum_i A[i,j]*feat2d[i,c] + b2[c] ----------
// g2 flat [j*25+c] == out_gcn in [25][6400] channel-major (the reference's .view mixing)
__global__ void gcn_agg(const float* __restrict__ q, const float* __restrict__ k,
                        const float* __restrict__ tthr, const float* __restrict__ dinv,
                        const float* __restrict__ feat2d, const float* __restrict__ b2,
                        float* __restrict__ g2out){
  int col = blockIdx.x*blockDim.y + threadIdx.y;
  int lane = threadIdx.x;
  float k0=k[col], k1=k[NP+col], k2=k[2*NP+col];
  float acc[25];
  #pragma unroll
  for(int c=0;c<25;c++) acc[c]=0.0f;
  for(int i=lane;i<NP;i+=32){
    float e = q[i]*k0 + q[NP+i]*k1 + q[2*NP+i]*k2;
    if((i==col) || (e>=tthr[i])){
      const float* fr = feat2d + i*25;
      #pragma unroll
      for(int c=0;c<25;c++) acc[c] += fr[c];
    }
  }
  for(int off=16; off>0; off>>=1){
    #pragma unroll
    for(int c=0;c<25;c++) acc[c] += __shfl_xor(acc[c],off);
  }
  if(lane==0){
    float dj = dinv[col];
    for(int c=0;c<25;c++) g2out[col*25+c] = dj*acc[c] + b2[c];
  }
}

extern "C" void kernel_launch(void* const* d_in, const int* in_sizes, int n_in,
                              void* d_out, int out_size, void* d_ws, size_t ws_size,
                              hipStream_t stream){
  // ---- input map (setup_inputs dict order, nested dicts flattened) ----
  const float* x       = (const float*)d_in[0];                       // [198][6400]
  const float *b1g=(const float*)d_in[1],  *b1b=(const float*)d_in[2],  *b1m=(const float*)d_in[3],  *b1v=(const float*)d_in[4];
  const float *con1w=(const float*)d_in[5],  *con1b=(const float*)d_in[6];
  const float *cond1w=(const float*)d_in[7], *cond1b=(const float*)d_in[8];
  const float *b2g=(const float*)d_in[9],  *b2b=(const float*)d_in[10], *b2m=(const float*)d_in[11], *b2v=(const float*)d_in[12];
  const float *con2w=(const float*)d_in[13], *con2b=(const float*)d_in[14];
  const float *cond2w=(const float*)d_in[15],*cond2b=(const float*)d_in[16];
  const float *b4g=(const float*)d_in[17], *b4b=(const float*)d_in[18], *b4m=(const float*)d_in[19], *b4v=(const float*)d_in[20];
  const float *qw=(const float*)d_in[21],  *qb=(const float*)d_in[22];
  const float *kw=(const float*)d_in[23],  *kb_=(const float*)d_in[24];
  const float *tw=(const float*)d_in[25],  *tb=(const float*)d_in[26];
  const float *g1w=(const float*)d_in[27], *g1b=(const float*)d_in[28];
  const float *g2w=(const float*)d_in[29], *g2b=(const float*)d_in[30];
  const float *bcg=(const float*)d_in[31], *bcb=(const float*)d_in[32], *bcm=(const float*)d_in[33], *bcv=(const float*)d_in[34];
  const float *con4w=(const float*)d_in[35], *con4b=(const float*)d_in[36];
  const float *cond4w=(const float*)d_in[37],*cond4b=(const float*)d_in[38];
  const float *b5g=(const float*)d_in[39], *b5b=(const float*)d_in[40], *b5m=(const float*)d_in[41], *b5v=(const float*)d_in[42];
  const float *con5w=(const float*)d_in[43], *con5b=(const float*)d_in[44];
  const float *cond5w=(const float*)d_in[45],*cond5b=(const float*)d_in[46];
  const float *con6w=(const float*)d_in[47], *con6b=(const float*)d_in[48];
  float* outp = (float*)d_out;                                        // [3][6400]

  // ---- workspace layout (floats); total ~3.96M floats ~15.8MB ----
  float* W = (float*)d_ws;
  float *W1f = W,          *b1f = W+9900;      // 50x198
  float *W2f = W+9950,     *b2f = W+22350;     // 50x248
  float *W4f = W+22400,    *b4f = W+24900;     // 50x50
  float *W5f = W+24950,    *b5f = W+28700;     // 50x75
  float *cat2  = W+28800;                      // [248][NP]: out1 rows 0-49, x rows 50-247
  float *tmp   = W+1616000;                    // [50][NP] scratch
  float *out2  = W+1936000;                    // [25][NP]
  float *cat4  = W+2096000;                    // [50][NP]: xx rows 0-24, out_gcn rows 25-49
  float *cat5  = W+2416000;                    // [75][NP]: out4/out5 rows 0-49, out2 rows 50-74
  float *qbuf  = W+2896000;                    // [3][NP]
  float *kbuf  = W+2915200;                    // [3][NP]
  float *xtb   = W+2934400;                    // [NP]
  float *tthr  = W+2940800;                    // [NP]
  float *dinv  = W+2947200;                    // [NP]
  float *sbuf  = W+2953600;                    // [NP]
  float *feat2d= W+2960000;                    // [NP][25]
  _Float16 *packA = (_Float16*)(W+3120000);    // <=16384 halfs (4 mtiles x 8 kchunks x 512)
  _Float16 *packB = (_Float16*)(W+3130000);    // <=1,638,400 halfs (8 kchunks x 400 ptiles x 512)

  dim3 wblk(32,4);                             // 4 waves / block
  dim3 wgrd(1600);                             // 6400 rows/cols, 4 per block

  auto packX = [&](const float* Xi, int K)->int{
    int nkc = (K+31)/32;
    int total = nkc*NPT*512;
    pack_x_f16<<<(total+255)/256,256,0,stream>>>(Xi,packB,K,nkc);
    return nkc;
  };
  auto gemm = [&](const float* Wf,const float* bs,int M,int K,int nkc,float* o,int act){
    int nmt = (M+15)/16;
    int totalw = nmt*nkc*512;
    pack_w_f16<<<(totalw+255)/256,256,0,stream>>>(Wf,packA,M,K,nkc);
    dim3 blk(32,4), grd(NPT/4, nmt);
    gemm_wmma<<<grd,blk,0,stream>>>(packA,packB,bs,o,M,nkc,act);
  };

  // 1) fold BN into 1x1 convs
  fold_bn_conv<<<1,64,0,stream>>>(con1w,con1b,b1g,b1b,b1m,b1v,50,198,W1f,b1f);
  fold_bn_conv<<<1,64,0,stream>>>(con2w,con2b,b2g,b2b,b2m,b2v,50,248,W2f,b2f);
  fold_bn_conv<<<1,64,0,stream>>>(con4w,con4b,bcg,bcb,bcm,bcv,50,50,W4f,b4f);
  fold_bn_conv<<<1,64,0,stream>>>(con5w,con5b,b5g,b5b,b5m,b5v,50,75,W5f,b5f);

  // 2) stage x into cat2 rows 50-247 (concat [out1, x])
  hipMemcpyAsync(cat2+50*NP, x, (size_t)198*NP*sizeof(float), hipMemcpyDeviceToDevice, stream);

  // 3) out1 = sig(dw5x5(sig(con1(BN(x)))))
  int nkc = packX(x,198);
  gemm(W1f,b1f,50,198,nkc, tmp,1);
  dwconv5x5_sig<<<(50*NP+255)/256,256,0,stream>>>(tmp,cond1w,cond1b,cat2,50,1);

  // 4) out2 = sig(grouped5x5(sig(con2(BN(cat2)))))
  nkc = packX(cat2,248);
  gemm(W2f,b2f,50,248,nkc, tmp,1);
  dwconv5x5_sig<<<(25*NP+255)/256,256,0,stream>>>(tmp,cond2w,cond2b,out2,25,2);
  hipMemcpyAsync(cat5+50*NP, out2, (size_t)25*NP*sizeof(float), hipMemcpyDeviceToDevice, stream);

  // 5) xx = BN(out2) into cat4 rows 0-24; q,k,xt via 1x1 convs (shared packed B)
  bn_elem<<<(25*NP+255)/256,256,0,stream>>>(out2,b4g,b4b,b4m,b4v,cat4,25);
  nkc = packX(cat4,25);
  gemm(qw,qb, 3,25,nkc, qbuf,0);
  gemm(kw,kb_,3,25,nkc, kbuf,0);
  gemm(tw,tb, 1,25,nkc, xtb, 0);

  // 6) attention thresholds (no 164MB att matrix; log-domain compare)
  row_stats<<<wgrd,wblk,0,stream>>>(qbuf,kbuf,tthr);
  col_deg  <<<wgrd,wblk,0,stream>>>(qbuf,kbuf,tthr,dinv);

  // 7) GCN: rank-1 layer1 scalar, then masked 25-wide aggregation
  col_sacc<<<wgrd,wblk,0,stream>>>(qbuf,kbuf,tthr,dinv,xtb,sbuf);
  feat2_build<<<(NP+255)/256,256,0,stream>>>(dinv,sbuf,g1w,g1b,g2w,feat2d);
  gcn_agg<<<wgrd,wblk,0,stream>>>(qbuf,kbuf,tthr,dinv,feat2d,g2b,cat4+25*NP);

  // 8) out4 = sig(dw5x5(sig(con4(BN(cat4)))))
  nkc = packX(cat4,50);
  gemm(W4f,b4f,50,50,nkc, tmp,1);
  dwconv5x5_sig<<<(50*NP+255)/256,256,0,stream>>>(tmp,cond4w,cond4b,cat5,50,1);

  // 9) out5 = sig(dw5x5(sig(con5(BN(cat5)))))   (cat5 rows 50-74 already hold out2)
  nkc = packX(cat5,75);
  gemm(W5f,b5f,50,75,nkc, tmp,1);
  dwconv5x5_sig<<<(50*NP+255)/256,256,0,stream>>>(tmp,cond5w,cond5b,cat5,50,1);

  // 10) out6 = con6(cat6)  (cat5 now = [out5, out2], no BN, no activation; repack)
  nkc = packX(cat5,75);
  gemm(con6w,con6b,3,75,nkc, outp,0);
}
